// Matching_52424370815581
// MI455X (gfx1250) — matrix operand order repaired and checked by reading, not compiled, not run on hardware
//
#include <hip/hip_runtime.h>
#include <hip/hip_bf16.h>

typedef __attribute__((ext_vector_type(16))) __bf16 v16bf;
typedef __attribute__((ext_vector_type(8)))  __bf16 v8bf;
typedef __attribute__((ext_vector_type(8)))  float  v8f;
typedef __attribute__((ext_vector_type(4)))  unsigned int v4u;
typedef __attribute__((ext_vector_type(8)))  int v8i_t;
typedef __attribute__((ext_vector_type(4)))  int v4i_t;

#define NPIX 4096
#define CCH  256
#define FDCH 128

#if defined(__has_builtin)
#  if __has_builtin(__builtin_amdgcn_tensor_load_to_lds) && __has_builtin(__builtin_amdgcn_s_wait_tensorcnt)
#    define USE_TDM 1
#  endif
#endif
#ifndef USE_TDM
#  define USE_TDM 0
#endif

// ---- Tensor Data Mover: 1-D contiguous bf16 tile -> LDS with 8-elem pad every 256 elems ----
// D# per CDNA5 ISA: group0 {count|lds_addr|global_addr|type=2},
// group1 {data_size=2B, pad_enable, pad_interval=128dw, pad_amount=4dw, dims}.
// This toolchain exposes the 6-arg builtin: (v4u, v8i, v4i, v4i, v8i, i32 cpol).
__device__ __forceinline__ void tdm_load_1d_padded(const void* gsrc, void* lds_dst, unsigned nelem) {
#if USE_TDM
  unsigned long long ga = (unsigned long long)(uintptr_t)gsrc;
  unsigned lz = (unsigned)(uintptr_t)lds_dst;           // low 32 bits = LDS byte offset
  v4u g0;
  g0.x = 1u;                                            // count=1, user mode
  g0.y = lz;                                            // lds_addr
  g0.z = (unsigned)(ga & 0xffffffffu);                  // global_addr[31:0]
  g0.w = (unsigned)((ga >> 32) & 0x01ffffffu) | (2u << 30); // global_addr[56:32] | type=2
  v8i_t g1;
  g1[0] = (int)((1u << 16) | (1u << 20) | (6u << 22) | (3u << 25)); // 2B data, pad_en, 128dw interval, 4dw pad
  g1[1] = (int)((nelem & 0xffffu) << 16);               // abar_addr=0 | tensor_dim0[15:0]
  g1[2] = (int)((nelem >> 16) | (1u << 16));            // tensor_dim0[31:16] | tensor_dim1=1 (lo)
  g1[3] = (int)((nelem & 0xffffu) << 16);               // tensor_dim1 hi=0 | tile_dim0=nelem
  g1[4] = 0;                                            // tile_dim1=0 (unused), tile_dim2=0
  g1[5] = (int)nelem;                                   // tensor_dim0_stride[31:0]
  g1[6] = 0;
  g1[7] = 0;
  v4i_t z4 = {0, 0, 0, 0};
  v8i_t z8;
#pragma unroll
  for (int i = 0; i < 8; ++i) z8[i] = 0;
  __builtin_amdgcn_tensor_load_to_lds(g0, g1, z4, z4, z8, 0);
#else
  (void)gsrc; (void)lds_dst; (void)nelem;
#endif
}
__device__ __forceinline__ void tdm_wait() {
#if USE_TDM
  __builtin_amdgcn_s_wait_tensorcnt(0);
#endif
}

__device__ __forceinline__ v8f wmma_bf16(v16bf a, v16bf b, v8f c) {
  return __builtin_amdgcn_wmma_f32_16x16x32_bf16(false, a, false, b, (short)0, c, false, false);
}

// A-fragment (16x32 bf16) from global f32 weights, row-major [M][K].
__device__ __forceinline__ v16bf load_a_global(const float* __restrict__ W, int row,
                                               int stride, int k0, int lane) {
  const float* p = W + (size_t)row * stride + k0 + ((lane & 16) ? 8 : 0);
  v16bf a;
#pragma unroll
  for (int j = 0; j < 8; ++j) a[j] = (__bf16)p[j];
#pragma unroll
  for (int j = 0; j < 8; ++j) a[j + 8] = (__bf16)p[j + 16];
  return a;
}

__device__ __forceinline__ v16bf load_a_global_guard(const float* __restrict__ W, int row,
                                                     int stride, int kmax, int k0, int lane) {
  int kb = k0 + ((lane & 16) ? 8 : 0);
  v16bf a;
#pragma unroll
  for (int j = 0; j < 16; ++j) {
    int k = kb + ((j < 8) ? j : (8 + j));
    a[j] = (k < kmax) ? (__bf16)W[(size_t)row * stride + k] : (__bf16)0.0f;
  }
  return a;
}

__device__ __forceinline__ v16bf load_a_lds(const __bf16* rowbase, int k0, int lane) {
  const __bf16* p = rowbase + k0 + ((lane & 16) ? 8 : 0);
  v8bf lo = *(const v8bf*)(p);
  v8bf hi = *(const v8bf*)(p + 16);
  v16bf a;
#pragma unroll
  for (int j = 0; j < 8; ++j) { a[j] = lo[j]; a[j + 8] = hi[j]; }
  return a;
}

// B-fragment (32x16): col=lane%16; lane<16 K=k0+{0..7,16..23}; lane>=16 K=k0+{8..15,24..31}.
__device__ __forceinline__ v16bf load_b_lds(const __bf16* rowbase, int k0, int lane) {
  const __bf16* p = rowbase + k0 + ((lane & 16) ? 8 : 0);
  v8bf lo = *(const v8bf*)(p);
  v8bf hi = *(const v8bf*)(p + 16);
  v16bf b;
#pragma unroll
  for (int j = 0; j < 8; ++j) { b[j] = lo[j]; b[j + 8] = hi[j]; }
  return b;
}

// 256x64 GEMM from LDS hid: mode 0 -> f32 [c][pix]; mode 1 -> bf16 [pix][c] packed stores.
__device__ __forceinline__ void gemm_256x64_from_lds(const __bf16* hid, int hidStride,
                                                     const float* __restrict__ W,
                                                     const float* __restrict__ bias,
                                                     int p0, int lane, int w,
                                                     float* __restrict__ outf,
                                                     __bf16* __restrict__ outb, int mode) {
  v8f acc[2][4];
#pragma unroll
  for (int mi = 0; mi < 2; ++mi) {
    int mb = w * 32 + mi * 16 + 8 * (lane >> 4);
#pragma unroll
    for (int nt = 0; nt < 4; ++nt)
#pragma unroll
      for (int r = 0; r < 8; ++r) acc[mi][nt][r] = bias[mb + r];
  }
  for (int ks = 0; ks < 8; ++ks) {
    int k0 = ks * 32;
    v16bf a0 = load_a_global(W, w * 32 + (lane & 15), 256, k0, lane);
    v16bf a1 = load_a_global(W, w * 32 + 16 + (lane & 15), 256, k0, lane);
#pragma unroll
    for (int nt = 0; nt < 4; ++nt) {
      v16bf bb = load_b_lds(hid + (size_t)(nt * 16 + (lane & 15)) * hidStride, k0, lane);
      acc[0][nt] = wmma_bf16(a0, bb, acc[0][nt]);
      acc[1][nt] = wmma_bf16(a1, bb, acc[1][nt]);
    }
  }
  if (mode) {
#pragma unroll
    for (int mi = 0; mi < 2; ++mi)
#pragma unroll
      for (int nt = 0; nt < 4; ++nt) {
        int mb = w * 32 + mi * 16 + 8 * (lane >> 4);
        int p = p0 + nt * 16 + (lane & 15);
        v8bf pk;
#pragma unroll
        for (int r = 0; r < 8; ++r) pk[r] = (__bf16)acc[mi][nt][r];
        *(v8bf*)(outb + (size_t)p * 256 + mb) = pk;    // 16B packed store
      }
  } else {
#pragma unroll
    for (int mi = 0; mi < 2; ++mi)
#pragma unroll
      for (int nt = 0; nt < 4; ++nt)
#pragma unroll
        for (int r = 0; r < 8; ++r) {
          int m = w * 32 + mi * 16 + 8 * (lane >> 4) + r;
          int p = p0 + nt * 16 + (lane & 15);
          outf[(size_t)m * NPIX + p] = acc[mi][nt][r];
        }
  }
}

// ---------------- fuser ----------------
__global__ __launch_bounds__(256) void fuser_kernel(
    const float* __restrict__ xin, const float* __restrict__ feat,
    const float* __restrict__ w1, const float* __restrict__ b1,
    const float* __restrict__ w2, const float* __restrict__ b2,
    float* __restrict__ outf, __bf16* __restrict__ outb, int mode) {
  extern __shared__ __bf16 smem_dyn[];
  __bf16* inb = smem_dyn;               // [64][392]
  __bf16* hid = smem_dyn + 64 * 392;    // [64][264]
  const int b = blockIdx.z, p0 = blockIdx.x * 64;
  const int tid = threadIdx.x, lane = tid & 31, w = tid >> 5;

  for (int idx = tid; idx < 384 * 64; idx += 256) {
    int c = idx >> 6, p = idx & 63;
    float v = (c < 256) ? xin[((size_t)b * CCH + c) * NPIX + p0 + p]
                        : feat[((size_t)b * FDCH + (c - 256)) * NPIX + p0 + p];
    inb[(size_t)p * 392 + c] = (__bf16)v;
  }
  __syncthreads();

  v8f acc[2][4];
#pragma unroll
  for (int mi = 0; mi < 2; ++mi) {
    int mb = w * 32 + mi * 16 + 8 * (lane >> 4);
#pragma unroll
    for (int nt = 0; nt < 4; ++nt)
#pragma unroll
      for (int r = 0; r < 8; ++r) acc[mi][nt][r] = b1[mb + r];
  }
  for (int ks = 0; ks < 12; ++ks) {
    int k0 = ks * 32;
    v16bf a0 = load_a_global(w1, w * 32 + (lane & 15), 384, k0, lane);
    v16bf a1 = load_a_global(w1, w * 32 + 16 + (lane & 15), 384, k0, lane);
#pragma unroll
    for (int nt = 0; nt < 4; ++nt) {
      v16bf bb = load_b_lds(inb + (size_t)(nt * 16 + (lane & 15)) * 392, k0, lane);
      acc[0][nt] = wmma_bf16(a0, bb, acc[0][nt]);
      acc[1][nt] = wmma_bf16(a1, bb, acc[1][nt]);
    }
  }
#pragma unroll
  for (int mi = 0; mi < 2; ++mi)
#pragma unroll
    for (int nt = 0; nt < 4; ++nt)
#pragma unroll
      for (int r = 0; r < 8; ++r) {
        float v = fmaxf(acc[mi][nt][r], 0.f);
        int p = nt * 16 + (lane & 15);
        int m = w * 32 + mi * 16 + 8 * (lane >> 4) + r;
        hid[(size_t)p * 264 + m] = (__bf16)v;
      }
  __syncthreads();

  gemm_256x64_from_lds(hid, 264, w2, b2, p0, lane, w,
                       outf + (size_t)b * CCH * NPIX,
                       outb + (size_t)b * NPIX * CCH, mode);
}

// ---------------- pos_gen -> bf16 V matrix [b][48][4096] ----------------
__global__ __launch_bounds__(256) void posgen_kernel(
    const float* __restrict__ feat2,
    const float* __restrict__ pw1, const float* __restrict__ pb1,
    const float* __restrict__ pw2, const float* __restrict__ pb2,
    __bf16* __restrict__ posb) {
  extern __shared__ __bf16 smem_dyn[];
  __bf16* inb = smem_dyn;               // [64][168]
  __bf16* hid = smem_dyn + 64 * 168;    // [64][136]
  const int b = blockIdx.z, iy = blockIdx.x, p0 = iy * 64;
  const int tid = threadIdx.x, lane = tid & 31, w = tid >> 5;
  const float gy = -1.f + 2.f * (float)iy / 63.f;

  for (int idx = tid; idx < 160 * 64; idx += 256) {
    int c = idx >> 6, p = idx & 63;
    float gx = -1.f + 2.f * (float)p / 63.f;
    float v;
    if (c == 0) v = gx + 1.f;
    else if (c == 1) v = gy + 1.f;
    else if (c < 130) v = feat2[((size_t)b * FDCH + (c - 2)) * NPIX + p0 + p];
    else v = 0.f;
    inb[(size_t)p * 168 + c] = (__bf16)v;
  }
  __syncthreads();

  v8f acc[4];
  {
    int mb = w * 16 + 8 * (lane >> 4);
#pragma unroll
    for (int nt = 0; nt < 4; ++nt)
#pragma unroll
      for (int r = 0; r < 8; ++r) acc[nt][r] = pb1[mb + r];
  }
  for (int ks = 0; ks < 5; ++ks) {
    int k0 = ks * 32;
    v16bf a = load_a_global_guard(pw1, w * 16 + (lane & 15), 130, 130, k0, lane);
#pragma unroll
    for (int nt = 0; nt < 4; ++nt) {
      v16bf bb = load_b_lds(inb + (size_t)(nt * 16 + (lane & 15)) * 168, k0, lane);
      acc[nt] = wmma_bf16(a, bb, acc[nt]);
    }
  }
#pragma unroll
  for (int nt = 0; nt < 4; ++nt)
#pragma unroll
    for (int r = 0; r < 8; ++r) {
      float v = fmaxf(acc[nt][r], 0.f);
      int p = nt * 16 + (lane & 15);
      int m = w * 16 + 8 * (lane >> 4) + r;
      hid[(size_t)p * 136 + m] = (__bf16)v;
    }
  __syncthreads();

  {
    int mi = w & 1, nt = w >> 1;
    v8f a2;
    int mb = mi * 16 + 8 * (lane >> 4);
#pragma unroll
    for (int r = 0; r < 8; ++r) a2[r] = pb2[mb + r];
    for (int ks = 0; ks < 4; ++ks) {
      int k0 = ks * 32;
      v16bf a = load_a_global(pw2, mi * 16 + (lane & 15), 128, k0, lane);
      v16bf bb = load_b_lds(hid + (size_t)(nt * 16 + (lane & 15)) * 136, k0, lane);
      a2 = wmma_bf16(a, bb, a2);
    }
#pragma unroll
    for (int r = 0; r < 8; ++r) {
      int ch = mi * 16 + 8 * (lane >> 4) + r;
      int p = p0 + nt * 16 + (lane & 15);
      posb[((size_t)b * 48 + ch) * NPIX + p] = (__bf16)fmaxf(a2[r], 0.f);
    }
  }
  for (int idx = tid; idx < 16 * 64; idx += 256) {
    int ch = 32 + (idx >> 6), p = idx & 63;
    float gx = -1.f + 2.f * (float)p / 63.f;
    float v = (ch == 32) ? gx : (ch == 33) ? gy : 0.f;
    posb[((size_t)b * 48 + ch) * NPIX + p0 + p] = (__bf16)v;
  }
}

// ---------------- conv_y: depthwise 3x3 + pointwise -> bf16 K matrix [b][key][256] ----------------
__global__ __launch_bounds__(256) void convy_kernel(
    const float* __restrict__ yf, const float* __restrict__ dww, const float* __restrict__ dwb,
    const float* __restrict__ pww, const float* __restrict__ pwb, __bf16* __restrict__ kfb) {
  extern __shared__ __bf16 smem_dyn[];   // [64][264]
  const int b = blockIdx.z, iy = blockIdx.x;
  const int tid = threadIdx.x, lane = tid & 31, w = tid >> 5;

  for (int idx = tid; idx < 256 * 64; idx += 256) {
    int c = idx >> 6, ix = idx & 63;
    float s = dwb[c];
#pragma unroll
    for (int ky = 0; ky < 3; ++ky) {
      int y2 = iy + ky - 1;
      if (y2 >= 0 && y2 < 64) {
#pragma unroll
        for (int kx = 0; kx < 3; ++kx) {
          int x2 = ix + kx - 1;
          if (x2 >= 0 && x2 < 64)
            s += yf[((size_t)b * CCH + c) * NPIX + y2 * 64 + x2] * dww[c * 9 + ky * 3 + kx];
        }
      }
    }
    smem_dyn[(size_t)ix * 264 + c] = (__bf16)s;
  }
  __syncthreads();
  gemm_256x64_from_lds(smem_dyn, 264, pww, pwb, iy * 64, lane, w,
                       (float*)nullptr, kfb + (size_t)b * NPIX * CCH, 1);
}

// ---------------- flash attention ----------------
__global__ __launch_bounds__(256) void attn_kernel(
    const __bf16* __restrict__ xfb, const __bf16* __restrict__ kfb,
    const __bf16* __restrict__ posb, const float* __restrict__ dvec,
    const float* __restrict__ dpos, float* __restrict__ out) {
  extern __shared__ __bf16 smem_dyn[];
  __bf16* qbuf = smem_dyn;                       // [128][264]
  __bf16* kbuf = smem_dyn + (size_t)128 * 264;   // [64][264]
  __bf16* pbuf = kbuf + (size_t)64 * 264;        // [8][16][64]
  const int b = blockIdx.z;
  const int q0blk = blockIdx.x * 128;
  const int tid = threadIdx.x, lane = tid & 31, w = tid >> 5;

  // stage Q tile: contiguous bf16 [128][256] -> LDS stride 264 via TDM padding
#if USE_TDM
  if (tid < 32) {
    tdm_load_1d_padded(xfb + ((size_t)b * NPIX + q0blk) * 256, qbuf, 128 * 256);
    tdm_wait();
  }
#else
  for (int idx = tid; idx < 128 * 256; idx += 256) {
    int q = idx >> 8, c = idx & 255;
    qbuf[(size_t)q * 264 + c] = xfb[((size_t)b * NPIX + q0blk + q) * 256 + c];
  }
#endif

  float mrow[8], lrow[8];
  v8f o[3];
#pragma unroll
  for (int r = 0; r < 8; ++r) { mrow[r] = -1e30f; lrow[r] = 0.f; }
#pragma unroll
  for (int t = 0; t < 3; ++t)
#pragma unroll
    for (int r = 0; r < 8; ++r) o[t][r] = 0.f;

  const __bf16* qrow = qbuf + (size_t)(w * 16 + (lane & 15)) * 264;
  __bf16* pwave = pbuf + (size_t)w * 1024;

  for (int kt = 0; kt < 65; ++kt) {
    __syncthreads();   // kbuf (and first-iter qbuf) ready/reusable
    if (kt < 64) {
#if USE_TDM
      if (tid < 32) {
        tdm_load_1d_padded(kfb + ((size_t)b * NPIX + kt * 64) * 256, kbuf, 64 * 256);
        tdm_wait();
      }
#else
      for (int idx = tid; idx < 64 * 256; idx += 256) {
        int j = idx >> 8, c = idx & 255;
        kbuf[(size_t)j * 264 + c] = kfb[((size_t)b * NPIX + kt * 64 + j) * 256 + c];
      }
#endif
      if (kt + 1 < 64)
        __builtin_prefetch(kfb + ((size_t)b * NPIX + (kt + 1) * 64) * 256 + tid * 64, 0, 1);
    } else {
      // dustbin tile: key row 0 = dustbin_vec, rest zero
      for (int idx = tid; idx < 64 * 256; idx += 256) {
        int j = idx >> 8, c = idx & 255;
        kbuf[(size_t)j * 264 + c] = (__bf16)((j == 0) ? dvec[c] : 0.f);
      }
    }
    __syncthreads();

    // scores: 16 queries x 64 keys, K=256
    v8f sacc[4];
#pragma unroll
    for (int nt = 0; nt < 4; ++nt)
#pragma unroll
      for (int r = 0; r < 8; ++r) sacc[nt][r] = 0.f;
    for (int ks = 0; ks < 8; ++ks) {
      int k0 = ks * 32;
      v16bf a = load_a_lds(qrow, k0, lane);
#pragma unroll
      for (int nt = 0; nt < 4; ++nt) {
        v16bf bb = load_b_lds(kbuf + (size_t)(nt * 16 + (lane & 15)) * 264, k0, lane);
        sacc[nt] = wmma_bf16(a, bb, sacc[nt]);
      }
    }
    float sc[4][8];
#pragma unroll
    for (int nt = 0; nt < 4; ++nt) {
      int key = kt * 64 + nt * 16 + (lane & 15);
      bool inv = key > 4096;
#pragma unroll
      for (int r = 0; r < 8; ++r) sc[nt][r] = inv ? -1e30f : sacc[nt][r] * 0.0625f;
    }

    float mnew[8], alpha[8];
#pragma unroll
    for (int r = 0; r < 8; ++r) {
      float mx = fmaxf(fmaxf(sc[0][r], sc[1][r]), fmaxf(sc[2][r], sc[3][r]));
      mx = fmaxf(mx, __shfl_xor(mx, 1, 32));
      mx = fmaxf(mx, __shfl_xor(mx, 2, 32));
      mx = fmaxf(mx, __shfl_xor(mx, 4, 32));
      mx = fmaxf(mx, __shfl_xor(mx, 8, 32));
      mnew[r] = fmaxf(mrow[r], mx);
      alpha[r] = __expf(mrow[r] - mnew[r]);
    }
#pragma unroll
    for (int nt = 0; nt < 4; ++nt)
#pragma unroll
      for (int r = 0; r < 8; ++r) {
        float p = __expf(sc[nt][r] - mnew[r]);
        sc[nt][r] = p;
        int qloc = (lane >> 4) * 8 + r;
        int key = nt * 16 + (lane & 15);
        pwave[(size_t)qloc * 64 + key] = (__bf16)p;
      }
#pragma unroll
    for (int r = 0; r < 8; ++r) {
      float ss = sc[0][r] + sc[1][r] + sc[2][r] + sc[3][r];
      ss += __shfl_xor(ss, 1, 32);
      ss += __shfl_xor(ss, 2, 32);
      ss += __shfl_xor(ss, 4, 32);
      ss += __shfl_xor(ss, 8, 32);
      lrow[r] = lrow[r] * alpha[r] + ss;
      mrow[r] = mnew[r];
    }
#pragma unroll
    for (int t = 0; t < 3; ++t)
#pragma unroll
      for (int r = 0; r < 8; ++r) o[t][r] *= alpha[r];

    // wave-private LDS turnaround (cross-lane within the wave)
    asm volatile("s_wait_dscnt 0" ::: "memory");

    // P @ V : M=16, K=64 keys, N=48 channels
    for (int ks2 = 0; ks2 < 2; ++ks2) {
      v16bf pa = load_a_lds(pwave + (size_t)(lane & 15) * 64, ks2 * 32, lane);
#pragma unroll
      for (int t = 0; t < 3; ++t) {
        int n = t * 16 + (lane & 15);
        int kb = ks2 * 32 + ((lane & 16) ? 8 : 0);
        v16bf vb;
        if (kt < 64) {
          const __bf16* vrow = posb + ((size_t)b * 48 + n) * NPIX + kt * 64 + kb;
          v8bf lo = *(const v8bf*)vrow;
          v8bf hi = *(const v8bf*)(vrow + 16);
#pragma unroll
          for (int j = 0; j < 8; ++j) { vb[j] = lo[j]; vb[j + 8] = hi[j]; }
        } else {
#pragma unroll
          for (int j = 0; j < 16; ++j) {
            int kk = kb + ((j < 8) ? j : (8 + j));
            vb[j] = (kk == 0 && n < 34) ? (__bf16)dpos[n] : (__bf16)0.f;
          }
        }
        o[t] = wmma_bf16(pa, vb, o[t]);
      }
    }
  }

  // epilogue: normalize, scatter [b][34][q]
#pragma unroll
  for (int t = 0; t < 3; ++t)
#pragma unroll
    for (int r = 0; r < 8; ++r) {
      int n = t * 16 + (lane & 15);
      if (n < 34) {
        int q = q0blk + w * 16 + 8 * (lane >> 4) + r;
        out[((size_t)b * 34 + n) * NPIX + q] = o[t][r] / lrow[r];
      }
    }
}

extern "C" void kernel_launch(void* const* d_in, const int* in_sizes, int n_in,
                              void* d_out, int out_size, void* d_ws, size_t ws_size,
                              hipStream_t stream) {
  (void)in_sizes; (void)n_in; (void)out_size; (void)ws_size;
  const float* x    = (const float*)d_in[0];
  const float* y    = (const float*)d_in[1];
  const float* f1   = (const float*)d_in[2];
  const float* f2   = (const float*)d_in[3];
  const float* pw1  = (const float*)d_in[4];
  const float* pb1  = (const float*)d_in[5];
  const float* pw2  = (const float*)d_in[6];
  const float* pb2  = (const float*)d_in[7];
  const float* fw1  = (const float*)d_in[8];
  const float* fb1  = (const float*)d_in[9];
  const float* fw2  = (const float*)d_in[10];
  const float* fb2  = (const float*)d_in[11];
  const float* dww  = (const float*)d_in[12];
  const float* dwb  = (const float*)d_in[13];
  const float* pww  = (const float*)d_in[14];
  const float* pwb  = (const float*)d_in[15];
  const float* dvec = (const float*)d_in[16];
  const float* dpos = (const float*)d_in[17];
  float* outp = (float*)d_out;

  float* ws = (float*)d_ws;
  const size_t IMGF = (size_t)8 * CCH * NPIX;         // yf (f32)
  float*  yf   = ws;
  __bf16* xfb  = (__bf16*)(ws + IMGF);                // [8][4096][256] bf16
  __bf16* kfb  = xfb + (size_t)8 * NPIX * CCH;        // [8][4096][256] bf16
  __bf16* posb = kfb + (size_t)8 * NPIX * CCH;        // [8][48][4096]  bf16

  dim3 blk(256);
  const size_t SM_FUSER = (size_t)(64 * 392 + 64 * 264) * 2;
  const size_t SM_POS   = (size_t)(64 * 168 + 64 * 136) * 2;
  const size_t SM_CONV  = (size_t)(64 * 264) * 2;
  const size_t SM_ATTN  = (size_t)(128 * 264 + 64 * 264 + 8 * 1024) * 2;

  // x path -> bf16 [q][c] for attention; y path -> f32 [c][p] for depthwise conv
  fuser_kernel<<<dim3(64, 1, 8), blk, SM_FUSER, stream>>>(x, f1, fw1, fb1, fw2, fb2,
                                                          yf /*unused*/, xfb, 1);
  fuser_kernel<<<dim3(64, 1, 8), blk, SM_FUSER, stream>>>(y, f2, fw1, fb1, fw2, fb2,
                                                          yf, xfb /*unused*/, 0);
  posgen_kernel<<<dim3(64, 1, 8), blk, SM_POS, stream>>>(f2, pw1, pb1, pw2, pb2, posb);
  convy_kernel<<<dim3(64, 1, 8), blk, SM_CONV, stream>>>(yf, dww, dwb, pww, pwb, kfb);
  attn_kernel<<<dim3(32, 1, 8), blk, SM_ATTN, stream>>>(xfb, kfb, posb, dvec, dpos, outp);
}
